// ZetaBlockEnhanced_63823214018762
// MI455X (gfx1250) — matrix operand order repaired and checked by reading, not compiled, not run on hardware
//
#include <hip/hip_runtime.h>
#include <hip/hip_bf16.h>
#include <stdint.h>

#define H_  16
#define B_  2
#define N_  2048
#define D_  1024
#define DH_ 64
#define BN_ (B_ * N_)

typedef __bf16 bf16_t;
typedef __attribute__((ext_vector_type(16))) __bf16 v16bf;
typedef __attribute__((ext_vector_type(8)))  __bf16 v8bf;
typedef __attribute__((ext_vector_type(8)))  float  v8f;

static __device__ __forceinline__ v8f wmma_bf16(v16bf a, v16bf b, v8f c) {
  return __builtin_amdgcn_wmma_f32_16x16x32_bf16(
      /*neg_a=*/false, a, /*neg_b=*/false, b,
      /*c_mod=*/(short)0, c, /*reuse_a=*/false, /*reuse_b=*/false);
}

// ---- CDNA5 async global->LDS copy (16B per lane), tracked by ASYNCcnt ----
static __device__ __forceinline__ void async_copy_b128(uint32_t lds_off,
                                                       const void* gaddr) {
  asm volatile("global_load_async_to_lds_b128 %0, %1, off"
               :: "v"(lds_off), "v"((unsigned long long)(uintptr_t)gaddr)
               : "memory");
}
static __device__ __forceinline__ void wait_async0() {
  asm volatile("s_wait_asynccnt 0x0" ::: "memory");
}

// A-fragment 16x32 bf16: lane<16 -> row=row0+lane, K = k0 + 0..15
//                        lane>=16 -> row=row0+lane-16, K = k0 + 16..31
static __device__ __forceinline__ v16bf load_frag_a(const bf16_t* base, int ld,
                                                    int row0, int k0, int lane) {
  int r  = row0 + (lane & 15);
  int kb = k0 + ((lane >> 4) << 4);
  const bf16_t* p = base + (size_t)r * ld + kb;
  v8bf lo = *(const v8bf*)(p);
  v8bf hi = *(const v8bf*)(p + 8);
  v16bf a;
#pragma unroll
  for (int i = 0; i < 8; ++i) { a[i] = lo[i]; a[i + 8] = hi[i]; }
  return a;
}

// B-fragment 32x16 bf16 from K-contiguous [col][k] storage:
//   lane<16 -> col=col0+lane, K=k0+0..15 ; lane>=16 -> col=col0+lane-16, K=k0+16..31
static __device__ __forceinline__ v16bf load_frag_b(const bf16_t* base, int ld,
                                                    int col0, int k0, int lane) {
  int c  = col0 + (lane & 15);
  int kb = k0 + ((lane >> 4) << 4);
  return *(const v16bf*)(base + (size_t)c * ld + kb);
}

__global__ __launch_bounds__(256) void f32_to_bf16_kernel(const float* __restrict__ in,
                                                          bf16_t* __restrict__ out, int n) {
  int i = blockIdx.x * blockDim.x + threadIdx.x;
  if (i < n) out[i] = (bf16_t)in[i];
}

// C[m, j] = sum_k A[m,k] * Wt[j,k] + bias[j]
// Block tile: M=128 (8 waves x 16 rows), N=64. Weight tile staged in LDS by
// async-to-LDS DMA, double-buffered across the K loop.
// outMode: 0 = bf16 row-major [M,Nn]; 1 = bf16 V-transpose (b,h,d,n); 2 = f32 row-major
__global__ __launch_bounds__(256) void gemm_nt_bf16_kernel(
    const bf16_t* __restrict__ A, const bf16_t* __restrict__ Wt,
    const float* __restrict__ bias, void* __restrict__ Out,
    int M, int Nn, int Kk, int outMode) {
  __shared__ __align__(64) bf16_t ldsB[2][64 * 32];   // [buf][col*32 + k], 4KB each

  int lane = threadIdx.x & 31;
  int wv   = threadIdx.x >> 5;                        // 0..7
  int tilesN = Nn >> 6;
  int m0 = (blockIdx.x / tilesN) * 128 + wv * 16;
  int j0 = (blockIdx.x % tilesN) * 64;

  // per-thread 16B slice of the 64x32 weight tile
  int tcol = threadIdx.x >> 2;                        // 0..63
  int tseg = (threadIdx.x & 3) << 3;                  // 0,8,16,24 (halves)
  const bf16_t* gB = Wt + (size_t)(j0 + tcol) * Kk + tseg;
  uint32_t ldsOff0 = (uint32_t)(uintptr_t)&ldsB[0][tcol * 32 + tseg];
  uint32_t ldsOff1 = (uint32_t)(uintptr_t)&ldsB[1][tcol * 32 + tseg];

  v8f acc[4];
#pragma unroll
  for (int g = 0; g < 4; ++g) { v8f z = {}; acc[g] = z; }

  async_copy_b128(ldsOff0, gB);                       // stage k0 = 0
  wait_async0();
  __syncthreads();

  for (int k0 = 0; k0 < Kk; k0 += 32) {
    int buf = (k0 >> 5) & 1;
    if (k0 + 32 < Kk)                                 // overlap next tile DMA
      async_copy_b128(buf ? ldsOff0 : ldsOff1, gB + k0 + 32);

    v16bf af = load_frag_a(A, Kk, m0, k0, lane);
    v16bf b0 = load_frag_b(ldsB[buf], 32,  0, 0, lane);
    v16bf b1 = load_frag_b(ldsB[buf], 32, 16, 0, lane);
    v16bf b2 = load_frag_b(ldsB[buf], 32, 32, 0, lane);
    v16bf b3 = load_frag_b(ldsB[buf], 32, 48, 0, lane);
    acc[0] = wmma_bf16(af, b0, acc[0]);
    acc[1] = wmma_bf16(af, b1, acc[1]);
    acc[2] = wmma_bf16(af, b2, acc[2]);
    acc[3] = wmma_bf16(af, b3, acc[3]);

    wait_async0();                                    // next tile landed
    __syncthreads();                                  // everyone done with old buf
  }

#pragma unroll
  for (int g = 0; g < 4; ++g) {
#pragma unroll
    for (int j = 0; j < 8; ++j) {
      int r = m0 + j + ((lane >> 4) << 3);
      int c = j0 + g * 16 + (lane & 15);
      float v = acc[g][j] + bias[c];
      if (outMode == 0) {
        ((bf16_t*)Out)[(size_t)r * Nn + c] = (bf16_t)v;
      } else if (outMode == 1) {
        int bb = r >> 11, n = r & (N_ - 1);
        int h = c >> 6, d = c & (DH_ - 1);
        ((bf16_t*)Out)[(((size_t)bb * H_ + h) * DH_ + d) * N_ + n] = (bf16_t)v;
      } else {
        ((float*)Out)[(size_t)r * Nn + c] = v;
      }
    }
  }
}

// Flash-style dual-kernel attention: one wave per 16 query rows of one (b,h).
__global__ __launch_bounds__(128) void attn_dual_kernel(
    const bf16_t* __restrict__ Q, const bf16_t* __restrict__ K,
    const bf16_t* __restrict__ Vt, const float* __restrict__ beta,
    const float* __restrict__ tvec, bf16_t* __restrict__ F) {
  __shared__ __align__(32) bf16_t ldsT[4][16 * 32];
  __shared__ __align__(32) bf16_t ldsS[4][16 * 32];

  int lane = threadIdx.x & 31;
  int w    = threadIdx.x >> 5;
  int gwid = blockIdx.x * 4 + w;
  int rowTiles = N_ >> 4;                       // 128
  int m0 = (gwid % rowTiles) << 4;
  int h  = (gwid / rowTiles) % H_;
  int b  = gwid / (rowTiles * H_);

  const bf16_t* Qb = Q + (size_t)b * N_ * D_ + h * DH_;
  const bf16_t* Kb = K + (size_t)b * N_ * D_ + h * DH_;
  const bf16_t* Vb = Vt + ((size_t)(b * H_ + h) * DH_) * N_;

  v16bf aQ0 = load_frag_a(Qb, D_, m0, 0, lane);
  v16bf aQ1 = load_frag_a(Qb, D_, m0, 32, lane);

  float invb = 1.0f / beta[h];
  float th = tvec[h];
  float t2 = th * th;
  const float scale = 0.125f;                   // 1/sqrt(64)

  v8f accT[4], accS[4];
#pragma unroll
  for (int g = 0; g < 4; ++g) { v8f z = {}; accT[g] = z; accS[g] = z; }
  float rmax[8], rsumT[8], rsumS[8];
#pragma unroll
  for (int j = 0; j < 8; ++j) { rmax[j] = -3.0e38f; rsumT[j] = 0.f; rsumS[j] = 0.f; }

  for (int mc = 0; mc < N_; mc += 32) {
    // ---- issue all global fetches for this chunk up front (K then V) ----
    v16bf bk00 = load_frag_b(Kb, D_, mc,      0,  lane);
    v16bf bk01 = load_frag_b(Kb, D_, mc,      32, lane);
    v16bf bk10 = load_frag_b(Kb, D_, mc + 16, 0,  lane);
    v16bf bk11 = load_frag_b(Kb, D_, mc + 16, 32, lane);
    v16bf bv[4];
#pragma unroll
    for (int g = 0; g < 4; ++g) bv[g] = load_frag_b(Vb, N_, g * 16, mc, lane);
    if (mc + 32 < N_) {
      __builtin_prefetch(Kb + (size_t)(mc + 32) * D_, 0, 3);
      __builtin_prefetch(Vb + (mc + 32), 0, 3);
    }

    // ---- two 16-col score tiles (S in C-layout) ----
    v8f z0 = {}, z1 = {};
    z0 = wmma_bf16(aQ0, bk00, z0);
    z0 = wmma_bf16(aQ1, bk01, z0);
    z1 = wmma_bf16(aQ0, bk10, z1);
    z1 = wmma_bf16(aQ1, bk11, z1);
    v8f s0 = z0 * scale;
    v8f s1 = z1 * scale;

    // ---- per-row online softmax + poisson weights, write P tiles to LDS ----
    // (V fetches above overlap with this VALU-heavy block)
#pragma unroll
    for (int j = 0; j < 8; ++j) {
      float a0 = s0[j], a1 = s1[j];
      float vmx = fmaxf(a0, a1);
#pragma unroll
      for (int off = 1; off < 16; off <<= 1) vmx = fmaxf(vmx, __shfl_xor(vmx, off, 16));
      float nm = fmaxf(rmax[j], vmx);
      float corr = __expf(rmax[j] - nm);
      rmax[j] = nm;
      rsumT[j] *= corr;
#pragma unroll
      for (int g = 0; g < 4; ++g) accT[g][j] *= corr;

      float p0 = __expf(a0 - nm), p1 = __expf(a1 - nm);
      float q0 = 1.0f + fabsf(a0) * invb;
      float q1 = 1.0f + fabsf(a1) * invb;
      float w0 = 1.0f / (q0 * q0 + t2);
      float w1 = 1.0f / (q1 * q1 + t2);
      float st = p0 + p1, ss = w0 + w1;
#pragma unroll
      for (int off = 1; off < 16; off <<= 1) {
        st += __shfl_xor(st, off, 16);
        ss += __shfl_xor(ss, off, 16);
      }
      rsumT[j] += st;
      rsumS[j] += ss;

      int row = j + ((lane >> 4) << 3);
      int cb  = lane & 15;
      ldsT[w][row * 32 + cb]      = (bf16_t)p0;
      ldsT[w][row * 32 + 16 + cb] = (bf16_t)p1;
      ldsS[w][row * 32 + cb]      = (bf16_t)w0;
      ldsS[w][row * 32 + 16 + cb] = (bf16_t)w1;
    }

    // ---- re-read P tiles as A-fragments (LDS transpose), accumulate P·V ----
    v16bf aPT = load_frag_a(ldsT[w], 32, 0, 0, lane);
    v16bf aPS = load_frag_a(ldsS[w], 32, 0, 0, lane);
#pragma unroll
    for (int g = 0; g < 4; ++g) {
      accT[g] = wmma_bf16(aPT, bv[g], accT[g]);
      accS[g] = wmma_bf16(aPS, bv[g], accS[g]);
    }
  }

  // ---- normalize both branches, f = 0.5*(tau + sigma) ----
#pragma unroll
  for (int g = 0; g < 4; ++g) {
#pragma unroll
    for (int j = 0; j < 8; ++j) {
      int row = j + ((lane >> 4) << 3);
      float f = 0.5f * (accT[g][j] / rsumT[j] + accS[g][j] / rsumS[j]);
      F[((size_t)b * N_ + m0 + row) * D_ + h * DH_ + g * 16 + (lane & 15)] = (bf16_t)f;
    }
  }
}

__global__ __launch_bounds__(256) void ln_residual_kernel(
    const float* __restrict__ x, const float* __restrict__ y,
    const float* __restrict__ lw, const float* __restrict__ lb,
    float* __restrict__ out) {
  int r = blockIdx.x;
  const float* yr = y + (size_t)r * D_;
  const float* xr = x + (size_t)r * D_;
  float s = 0.f, s2 = 0.f;
  for (int c = threadIdx.x; c < D_; c += 256) {
    float v = yr[c];
    s += v;
    s2 += v * v;
  }
  __shared__ float red0[256], red1[256];
  red0[threadIdx.x] = s;
  red1[threadIdx.x] = s2;
  __syncthreads();
  for (int off = 128; off > 0; off >>= 1) {
    if (threadIdx.x < (unsigned)off) {
      red0[threadIdx.x] += red0[threadIdx.x + off];
      red1[threadIdx.x] += red1[threadIdx.x + off];
    }
    __syncthreads();
  }
  float mu  = red0[0] * (1.0f / D_);
  float var = red1[0] * (1.0f / D_) - mu * mu;
  float rs  = rsqrtf(var + 1e-5f);
  for (int c = threadIdx.x; c < D_; c += 256) {
    float v = (yr[c] - mu) * rs;
    out[(size_t)r * D_ + c] = xr[c] + lw[c] * v + lb[c];
  }
}

extern "C" void kernel_launch(void* const* d_in, const int* in_sizes, int n_in,
                              void* d_out, int out_size, void* d_ws, size_t ws_size,
                              hipStream_t stream) {
  (void)in_sizes; (void)n_in; (void)out_size; (void)ws_size;
  const float* x    = (const float*)d_in[0];
  const float* Wq   = (const float*)d_in[1];
  const float* bq   = (const float*)d_in[2];
  const float* Wk   = (const float*)d_in[3];
  const float* bk   = (const float*)d_in[4];
  const float* Wv   = (const float*)d_in[5];
  const float* bv   = (const float*)d_in[6];
  const float* Wo   = (const float*)d_in[7];
  const float* bo   = (const float*)d_in[8];
  const float* beta = (const float*)d_in[9];
  const float* tv   = (const float*)d_in[10];
  const float* lw   = (const float*)d_in[11];
  const float* lb   = (const float*)d_in[12];
  float* out = (float*)d_out;

  char* ws = (char*)d_ws;
  const size_t MB = 1u << 20;
  bf16_t* xb  = (bf16_t*)(ws + 0 * MB);    // 8 MB  : bf16(x)  [BN, D]
  bf16_t* Wqb = (bf16_t*)(ws + 8 * MB);    // 2 MB
  bf16_t* Wkb = (bf16_t*)(ws + 10 * MB);   // 2 MB
  bf16_t* Wvb = (bf16_t*)(ws + 12 * MB);   // 2 MB
  bf16_t* Wob = (bf16_t*)(ws + 14 * MB);   // 2 MB
  bf16_t* Qb  = (bf16_t*)(ws + 16 * MB);   // 8 MB  : [BN, D]
  bf16_t* Kb  = (bf16_t*)(ws + 24 * MB);   // 8 MB  : [BN, D]
  bf16_t* Vtb = (bf16_t*)(ws + 32 * MB);   // 8 MB  : (b,h,d,n)
  bf16_t* Fb  = (bf16_t*)(ws + 40 * MB);   // 8 MB  : [BN, D]
  float*  Yb  = (float*)(ws + 48 * MB);    // 16 MB : [BN, D] fp32

  int nx = BN_ * D_;
  int nw = D_ * D_;
  f32_to_bf16_kernel<<<nx / 256, 256, 0, stream>>>(x, xb, nx);
  f32_to_bf16_kernel<<<nw / 256, 256, 0, stream>>>(Wq, Wqb, nw);
  f32_to_bf16_kernel<<<nw / 256, 256, 0, stream>>>(Wk, Wkb, nw);
  f32_to_bf16_kernel<<<nw / 256, 256, 0, stream>>>(Wv, Wvb, nw);
  f32_to_bf16_kernel<<<nw / 256, 256, 0, stream>>>(Wo, Wob, nw);

  int gemmBlocks = (BN_ / 128) * (D_ / 64);  // 128x64 tile per 256-thread block
  gemm_nt_bf16_kernel<<<gemmBlocks, 256, 0, stream>>>(xb, Wqb, bq, (void*)Qb, BN_, D_, D_, 0);
  gemm_nt_bf16_kernel<<<gemmBlocks, 256, 0, stream>>>(xb, Wkb, bk, (void*)Kb, BN_, D_, D_, 0);
  gemm_nt_bf16_kernel<<<gemmBlocks, 256, 0, stream>>>(xb, Wvb, bv, (void*)Vtb, BN_, D_, D_, 1);

  attn_dual_kernel<<<(B_ * H_ * (N_ / 16)) / 4, 128, 0, stream>>>(Qb, Kb, Vtb, beta, tv, Fb);

  gemm_nt_bf16_kernel<<<gemmBlocks, 256, 0, stream>>>(Fb, Wob, bo, (void*)Yb, BN_, D_, D_, 2);

  ln_residual_kernel<<<BN_, 256, 0, stream>>>(x, Yb, lw, lb, out);
}
